// CPA_432_56229711839617
// MI455X (gfx1250) — compile-verified, matrix operand order checked by reading, not run on hardware
//
#include <hip/hip_runtime.h>

// ---------------------------------------------------------------------------
// CDNA5 (gfx1250) implementation: bf16 WMMA GEMMs (fp32 accumulate) for all
// matrix products; fp32 VALU for LayerNorm / softmax reductions.
// GEMM: 32x32 register tile per wave -> 4 v_wmma per K-step, 2 b128 loads/wmma.
// ---------------------------------------------------------------------------

typedef __bf16 bf16_t;
typedef __attribute__((ext_vector_type(16))) __bf16 v16bf;
typedef __attribute__((ext_vector_type(8)))  __bf16 v8bf;
typedef __attribute__((ext_vector_type(8)))  float  v8f;

#define DEV __device__ __forceinline__

// Load one 16x32 bf16 WMMA fragment (A-layout; also used for B via the
// "BT = [N,K] row-major" convention). Per the CDNA5 ISA 16-bit A layout:
// lanes 0-15 hold row M=lane, K in {0..7} U {16..23}; lanes 16-31 hold the
// same rows with K offset +8. Both K chunks are 16B-contiguous -> 2x b128.
DEV v16bf load_frag16(const bf16_t* __restrict__ base, int ld, int r0, int k0) {
  int lane = threadIdx.x & 31;
  const bf16_t* p = base + (long)(r0 + (lane & 15)) * ld + (k0 + ((lane >> 4) << 3));
  v8bf lo = *(const v8bf*)p;         // K = kk .. kk+7
  v8bf hi = *(const v8bf*)(p + 16);  // K = kk+16 .. kk+23
  return __builtin_shufflevector(lo, hi, 0,1,2,3,4,5,6,7,8,9,10,11,12,13,14,15);
}

DEV v8f wmma_bf16(v16bf a, v16bf b, v8f c) {
  return __builtin_amdgcn_wmma_f32_16x16x32_bf16(false, a, false, b, (short)0, c, false, false);
}

enum { EP_F32 = 0, EP_F32_RES = 1, EP_BF16_GELU = 2, EP_BF16 = 3, EP_BF16_T = 4 };

// Generic batched wave-tile GEMM: 32x32 output tile per wave,
// C = A[M,K](bf16) * BT[N,K]^T(bf16). Grid = (N/32, M/32, batch).
// Batch index z = b*Hb + h with independent per-b / per-h strides so per-head
// attention GEMMs can scatter into the concatenated activation buffer.
template<int MODE>
__global__ void __launch_bounds__(32)
k_gemm(const bf16_t* __restrict__ A, int lda, long sAb, long sAh,
       const bf16_t* __restrict__ BT, int ldb, long sBb, long sBh,
       const float* __restrict__ bias,
       float* __restrict__ Cf, bf16_t* __restrict__ Cb, int ldc, long sCb, long sCh,
       const float* __restrict__ R, int ldr, long sRb, long sRh,
       int Hb, int K)
{
  int z = blockIdx.z, b = z / Hb, h = z % Hb;
  A  += (long)b * sAb + (long)h * sAh;
  BT += (long)b * sBb + (long)h * sBh;
  long co = (long)b * sCb + (long)h * sCh;
  long ro = (long)b * sRb + (long)h * sRh;
  int n0 = blockIdx.x << 5, m0 = blockIdx.y << 5;

  v8f acc00 = {0.f,0.f,0.f,0.f,0.f,0.f,0.f,0.f};
  v8f acc01 = acc00, acc10 = acc00, acc11 = acc00;
  for (int k = 0; k < K; k += 32) {
    v16bf af0 = load_frag16(A,  lda, m0,      k);
    v16bf af1 = load_frag16(A,  lda, m0 + 16, k);
    v16bf bf0 = load_frag16(BT, ldb, n0,      k);
    v16bf bf1 = load_frag16(BT, ldb, n0 + 16, k);
    acc00 = wmma_bf16(af0, bf0, acc00);
    acc01 = wmma_bf16(af0, bf1, acc01);
    acc10 = wmma_bf16(af1, bf0, acc10);
    acc11 = wmma_bf16(af1, bf1, acc11);
  }

  // C/D layout: element i -> row +i + (lane>=16 ? 8 : 0), col +(lane&15)
  int lane = threadIdx.x & 31;
  v8f accs[2][2] = {{acc00, acc01}, {acc10, acc11}};
#pragma unroll
  for (int mi = 0; mi < 2; ++mi) {
#pragma unroll
    for (int ni = 0; ni < 2; ++ni) {
      v8f acc = accs[mi][ni];
      int n  = n0 + ni * 16 + (lane & 15);
      int mb = m0 + mi * 16 + ((lane >> 4) << 3);
      float bv = (MODE == EP_F32 || MODE == EP_F32_RES || MODE == EP_BF16_GELU) ? bias[n] : 0.f;
#pragma unroll
      for (int i = 0; i < 8; ++i) {
        int m = mb + i;
        float v = acc[i] + bv;
        if      (MODE == EP_F32)      Cf[co + (long)m * ldc + n] = v;
        else if (MODE == EP_F32_RES)  Cf[co + (long)m * ldc + n] = v + R[ro + (long)m * ldr + n];
        else if (MODE == EP_BF16_GELU)
          Cb[co + (long)m * ldc + n] = (bf16_t)(0.5f * v * (1.f + erff(v * 0.70710678118f)));
        else if (MODE == EP_BF16)     Cb[co + (long)m * ldc + n] = (bf16_t)v;
        else /* EP_BF16_T */          Cb[co + (long)n * ldc + m] = (bf16_t)v;  // store C^T
      }
    }
  }
}

// LayerNorm over D=128, one wave per token (4 floats/lane, shfl_xor reduce).
// Optionally adds positional encoding; emits bf16 GEMM operands.
__global__ void k_ln(const float* __restrict__ x, const float* __restrict__ w,
                     const float* __restrict__ bvec, const float* __restrict__ pe,
                     bf16_t* __restrict__ out_pe, bf16_t* __restrict__ out_plain,
                     int ntok, long mtot)
{
  long row = (long)blockIdx.x * (blockDim.x >> 5) + (threadIdx.x >> 5);
  if (row >= mtot) return;
  int lane = threadIdx.x & 31;
  const float4 v = *(const float4*)(x + row * 128 + lane * 4);
  float s = v.x + v.y + v.z + v.w;
#pragma unroll
  for (int o = 16; o > 0; o >>= 1) s += __shfl_xor(s, o, 32);
  float mean = s * 0.0078125f;
  float dx[4] = {v.x - mean, v.y - mean, v.z - mean, v.w - mean};
  float s2 = dx[0]*dx[0] + dx[1]*dx[1] + dx[2]*dx[2] + dx[3]*dx[3];
#pragma unroll
  for (int o = 16; o > 0; o >>= 1) s2 += __shfl_xor(s2, o, 32);
  float rstd = rsqrtf(s2 * 0.0078125f + 1e-5f);
  int tib = (int)(row % ntok);
#pragma unroll
  for (int j = 0; j < 4; ++j) {
    int d = lane * 4 + j;
    float y = dx[j] * rstd * w[d] + bvec[d];
    if (out_plain) out_plain[row * 128 + d] = (bf16_t)y;
    if (out_pe)    out_pe[row * 128 + d]    = (bf16_t)(y + pe[(long)tib * 128 + d]);
  }
}

// Softmax over the token axis (per batch, per channel), online max+sum.
// 1024 threads/block: 8 token-groups x 128 channels; partials combined in LDS.
// All global accesses coalesced across the 128 contiguous channels.
__global__ void __launch_bounds__(1024)
k_colsoftmax(const float* __restrict__ q, bf16_t* __restrict__ qs, int ntok)
{
  const int G = 8;
  int b = blockIdx.x;
  int t = threadIdx.x, d = t & 127, g = t >> 7;
  int per = ntok / G;
  const float* base = q + (long)b * ntok * 128 + d;
  float lm = -3.4e38f, ls = 0.f;
  for (int j = 0; j < per; ++j) {
    float x = base[(long)(g * per + j) * 128];
    float nm = fmaxf(lm, x);
    ls = ls * __expf(lm - nm) + __expf(x - nm);
    lm = nm;
  }
  __shared__ float smax[1024], ssum[1024];
  smax[t] = lm; ssum[t] = ls;
  __syncthreads();
  float m = -3.4e38f;
#pragma unroll
  for (int gg = 0; gg < G; ++gg) m = fmaxf(m, smax[gg * 128 + d]);
  float s = 0.f;
#pragma unroll
  for (int gg = 0; gg < G; ++gg) s += ssum[gg * 128 + d] * __expf(smax[gg * 128 + d] - m);
  float inv = 1.f / s;
  bf16_t* ob = qs + (long)b * ntok * 128 + d;
  for (int j = 0; j < per; ++j) {
    long idx = (long)(g * per + j) * 128;
    ob[idx] = (bf16_t)(__expf(base[idx] - m) * inv);
  }
}

// Softmax over the 64-channel head axis, fused with a head-transpose so the
// result is directly the A-operand (ksT[b,h,d,n]) of the context GEMM.
__global__ void k_rowsoftmax_T(const float* __restrict__ kin, bf16_t* __restrict__ ksT, int nk)
{
  long tok = blockIdx.x;  // over B*nk, blockDim.x == 128 (one token)
  int b = (int)(tok / nk), n = (int)(tok % nk);
  int t = threadIdx.x, d = t & 63, h = t >> 6;
  float v = kin[tok * 128 + t];
  __shared__ float red[128];
  red[t] = v; __syncthreads();
  for (int o = 32; o > 0; o >>= 1) { if (d < o) red[t] = fmaxf(red[t], red[t + o]); __syncthreads(); }
  float m = red[t & 64];
  __syncthreads();
  float e = __expf(v - m);
  red[t] = e; __syncthreads();
  for (int o = 32; o > 0; o >>= 1) { if (d < o) red[t] += red[t + o]; __syncthreads(); }
  float s = red[t & 64];
  ksT[(((long)b * 2 + h) * 64 + d) * nk + n] = (bf16_t)(e / s);
}

// v [B,Nk,128] f32 -> vT [B,H,64,Nk] bf16 (B-operand of the context GEMM).
__global__ void k_headT(const float* __restrict__ vin, bf16_t* __restrict__ vT, int nk, long total)
{
  long i = (long)blockIdx.x * blockDim.x + threadIdx.x;
  if (i >= total) return;
  int c = (int)(i & 127); long tok = i >> 7;
  int b = (int)(tok / nk), n = (int)(tok % nk);
  int h = c >> 6, d = c & 63;
  vT[(((long)b * 2 + h) * 64 + d) * nk + n] = (bf16_t)vin[i];
}

// W [K,N] f32 -> WT [N,K] bf16 (weights are tiny; one-shot).
__global__ void k_wT(const float* __restrict__ W, bf16_t* __restrict__ WT, int K, int N)
{
  int i = blockIdx.x * 256 + threadIdx.x;
  if (i >= K * N) return;
  int k = i / N, n = i % N;
  WT[(long)n * K + k] = (bf16_t)W[i];
}

extern "C" void kernel_launch(void* const* d_in, const int* in_sizes, int n_in,
                              void* d_out, int out_size, void* d_ws, size_t ws_size,
                              hipStream_t stream)
{
  (void)in_sizes; (void)n_in; (void)out_size; (void)ws_size;
  const int B = 128, N2 = 1024, N3 = 256, N4 = 64, D = 128, MLP = 512, H = 2;
  const long M2 = (long)B * N2, M3 = (long)B * N3, M4 = (long)B * N4;

  const float* f2    = (const float*)d_in[0];
  const float* f3    = (const float*)d_in[1];
  const float* f4    = (const float*)d_in[2];
  const float* pe2   = (const float*)d_in[3];
  const float* pe3   = (const float*)d_in[4];
  const float* pe4   = (const float*)d_in[5];
  const float* ln1_w = (const float*)d_in[6];  const float* ln1_b = (const float*)d_in[7];
  const float* ln2_w = (const float*)d_in[8];  const float* ln2_b = (const float*)d_in[9];
  const float* ln3_w = (const float*)d_in[10]; const float* ln3_b = (const float*)d_in[11];
  const float* ln4_w = (const float*)d_in[12]; const float* ln4_b = (const float*)d_in[13];
  const float* q1_w  = (const float*)d_in[14]; const float* q1_b  = (const float*)d_in[15];
  const float* k1_w  = (const float*)d_in[16]; const float* k1_b  = (const float*)d_in[17];
  const float* v1_w  = (const float*)d_in[18]; const float* v1_b  = (const float*)d_in[19];
  const float* q2_w  = (const float*)d_in[20]; const float* q2_b  = (const float*)d_in[21];
  const float* k2_w  = (const float*)d_in[22]; const float* k2_b  = (const float*)d_in[23];
  const float* v2_w  = (const float*)d_in[24]; const float* v2_b  = (const float*)d_in[25];
  const float* rp_w  = (const float*)d_in[26]; const float* rp_b  = (const float*)d_in[27];
  const float* ff1_w = (const float*)d_in[28]; const float* ff1_b = (const float*)d_in[29];
  const float* ff2_w = (const float*)d_in[30]; const float* ff2_b = (const float*)d_in[31];

  char* base = (char*)d_ws;
  size_t off = 0;
  auto alloc = [&](size_t bytes) -> char* {
    char* p = base + off; off += (bytes + 255) & ~(size_t)255; return p;
  };

  // Transposed bf16 weights
  bf16_t* q1T  = (bf16_t*)alloc((size_t)D * D * 2);
  bf16_t* q2T  = (bf16_t*)alloc((size_t)D * D * 2);
  bf16_t* k1T  = (bf16_t*)alloc((size_t)D * D * 2);
  bf16_t* k2T  = (bf16_t*)alloc((size_t)D * D * 2);
  bf16_t* v1T  = (bf16_t*)alloc((size_t)D * D * 2);
  bf16_t* v2T  = (bf16_t*)alloc((size_t)D * D * 2);
  bf16_t* rpT  = (bf16_t*)alloc((size_t)2 * D * D * 2);   // [128, 256]
  bf16_t* ff1T = (bf16_t*)alloc((size_t)D * MLP * 2);     // [512, 128]
  bf16_t* ff2T = (bf16_t*)alloc((size_t)MLP * D * 2);     // [128, 512]

  // Activations
  bf16_t* t2pe = (bf16_t*)alloc((size_t)M2 * D * 2);      // LN(f2)+pe2; reused later as xln
  bf16_t* t3pe = (bf16_t*)alloc((size_t)M3 * D * 2);
  bf16_t* t3   = (bf16_t*)alloc((size_t)M3 * D * 2);
  bf16_t* t4pe = (bf16_t*)alloc((size_t)M4 * D * 2);
  bf16_t* t4   = (bf16_t*)alloc((size_t)M4 * D * 2);
  float*  q1f  = (float*) alloc((size_t)M2 * D * 4);      // reused later as x
  float*  q2f  = (float*) alloc((size_t)M2 * D * 4);
  float*  k3f  = (float*) alloc((size_t)M3 * D * 4);
  float*  v3f  = (float*) alloc((size_t)M3 * D * 4);
  float*  k4f  = (float*) alloc((size_t)M4 * D * 4);
  float*  v4f  = (float*) alloc((size_t)M4 * D * 4);
  bf16_t* qs1  = (bf16_t*)alloc((size_t)M2 * D * 2);
  bf16_t* qs2  = (bf16_t*)alloc((size_t)M2 * D * 2);
  bf16_t* ksT3 = (bf16_t*)alloc((size_t)B * H * 64 * N3 * 2);
  bf16_t* vT3  = (bf16_t*)alloc((size_t)B * H * 64 * N3 * 2);
  bf16_t* ksT4 = (bf16_t*)alloc((size_t)B * H * 64 * N4 * 2);
  bf16_t* vT4  = (bf16_t*)alloc((size_t)B * H * 64 * N4 * 2);
  bf16_t* ctxT3= (bf16_t*)alloc((size_t)B * H * 64 * 64 * 2);
  bf16_t* ctxT4= (bf16_t*)alloc((size_t)B * H * 64 * 64 * 2);
  bf16_t* acat = (bf16_t*)alloc((size_t)M2 * 2 * D * 2);  // [M2, 256]
  bf16_t* hh   = (bf16_t*)alloc((size_t)M2 * MLP * 2);    // GELU(ff1) output
  float*  xbuf = q1f;   // x = f2 + attn   (q1f dead after col-softmax)
  bf16_t* xln  = t2pe;  // LN4(x)          (t2pe dead after q projections)

  // --- weight transposes (tiny) ---
  k_wT<<<dim3((D * D + 255) / 256),   256, 0, stream>>>(q1_w, q1T, D, D);
  k_wT<<<dim3((D * D + 255) / 256),   256, 0, stream>>>(q2_w, q2T, D, D);
  k_wT<<<dim3((D * D + 255) / 256),   256, 0, stream>>>(k1_w, k1T, D, D);
  k_wT<<<dim3((D * D + 255) / 256),   256, 0, stream>>>(k2_w, k2T, D, D);
  k_wT<<<dim3((D * D + 255) / 256),   256, 0, stream>>>(v1_w, v1T, D, D);
  k_wT<<<dim3((D * D + 255) / 256),   256, 0, stream>>>(v2_w, v2T, D, D);
  k_wT<<<dim3((2 * D * D + 255) / 256), 256, 0, stream>>>(rp_w,  rpT,  2 * D, D);
  k_wT<<<dim3((D * MLP + 255) / 256),  256, 0, stream>>>(ff1_w, ff1T, D, MLP);
  k_wT<<<dim3((MLP * D + 255) / 256),  256, 0, stream>>>(ff2_w, ff2T, MLP, D);

  // --- LayerNorm (+PE) -> bf16 operands ---
  k_ln<<<dim3(M2 / 8), 256, 0, stream>>>(f2, ln1_w, ln1_b, pe2, t2pe, nullptr, N2, M2);
  k_ln<<<dim3(M3 / 8), 256, 0, stream>>>(f3, ln2_w, ln2_b, pe3, t3pe, t3,      N3, M3);
  k_ln<<<dim3(M4 / 8), 256, 0, stream>>>(f4, ln3_w, ln3_b, pe4, t4pe, t4,      N4, M4);

  // --- projections (WMMA); grid = (N/32, M/32) ---
  k_gemm<EP_F32><<<dim3(D / 32, M2 / 32, 1), 32, 0, stream>>>(
      t2pe, D, 0, 0, q1T, D, 0, 0, q1_b, q1f, nullptr, D, 0, 0, nullptr, 0, 0, 0, 1, D);
  k_gemm<EP_F32><<<dim3(D / 32, M2 / 32, 1), 32, 0, stream>>>(
      t2pe, D, 0, 0, q2T, D, 0, 0, q2_b, q2f, nullptr, D, 0, 0, nullptr, 0, 0, 0, 1, D);
  k_gemm<EP_F32><<<dim3(D / 32, M3 / 32, 1), 32, 0, stream>>>(
      t3pe, D, 0, 0, k1T, D, 0, 0, k1_b, k3f, nullptr, D, 0, 0, nullptr, 0, 0, 0, 1, D);
  k_gemm<EP_F32><<<dim3(D / 32, M3 / 32, 1), 32, 0, stream>>>(
      t3,   D, 0, 0, v1T, D, 0, 0, v1_b, v3f, nullptr, D, 0, 0, nullptr, 0, 0, 0, 1, D);
  k_gemm<EP_F32><<<dim3(D / 32, M4 / 32, 1), 32, 0, stream>>>(
      t4pe, D, 0, 0, k2T, D, 0, 0, k2_b, k4f, nullptr, D, 0, 0, nullptr, 0, 0, 0, 1, D);
  k_gemm<EP_F32><<<dim3(D / 32, M4 / 32, 1), 32, 0, stream>>>(
      t4,   D, 0, 0, v2T, D, 0, 0, v2_b, v4f, nullptr, D, 0, 0, nullptr, 0, 0, 0, 1, D);

  // --- softmaxes ---
  k_colsoftmax<<<dim3(B), 1024, 0, stream>>>(q1f, qs1, N2);
  k_colsoftmax<<<dim3(B), 1024, 0, stream>>>(q2f, qs2, N2);
  k_rowsoftmax_T<<<dim3(B * N3), 128, 0, stream>>>(k3f, ksT3, N3);
  k_rowsoftmax_T<<<dim3(B * N4), 128, 0, stream>>>(k4f, ksT4, N4);
  k_headT<<<dim3((unsigned)((M3 * 128 + 255) / 256)), 256, 0, stream>>>(v3f, vT3, N3, M3 * 128);
  k_headT<<<dim3((unsigned)((M4 * 128 + 255) / 256)), 256, 0, stream>>>(v4f, vT4, N4, M4 * 128);

  // --- context GEMMs: ctx[d,e] = sum_n ks[n,d] v[n,e]; store ctx^T (bf16) ---
  k_gemm<EP_BF16_T><<<dim3(2, 2, B * H), 32, 0, stream>>>(
      ksT3, N3, (long)H * 64 * N3, (long)64 * N3,
      vT3,  N3, (long)H * 64 * N3, (long)64 * N3, nullptr,
      nullptr, ctxT3, 64, (long)H * 64 * 64, (long)64 * 64,
      nullptr, 0, 0, 0, H, N3);
  k_gemm<EP_BF16_T><<<dim3(2, 2, B * H), 32, 0, stream>>>(
      ksT4, N4, (long)H * 64 * N4, (long)64 * N4,
      vT4,  N4, (long)H * 64 * N4, (long)64 * N4, nullptr,
      nullptr, ctxT4, 64, (long)H * 64 * 64, (long)64 * 64,
      nullptr, 0, 0, 0, H, N4);

  // --- attn out: out[n,e] = sum_d qs[n,d] ctx[d,e]; scatter into acat cols ---
  k_gemm<EP_BF16><<<dim3(2, N2 / 32, B * H), 32, 0, stream>>>(
      qs1, D, (long)N2 * D, 64,
      ctxT3, 64, (long)H * 64 * 64, (long)64 * 64, nullptr,
      nullptr, acat, 2 * D, (long)N2 * 2 * D, 64,
      nullptr, 0, 0, 0, H, 64);
  k_gemm<EP_BF16><<<dim3(2, N2 / 32, B * H), 32, 0, stream>>>(
      qs2, D, (long)N2 * D, 64,
      ctxT4, 64, (long)H * 64 * 64, (long)64 * 64, nullptr,
      nullptr, acat + D, 2 * D, (long)N2 * 2 * D, 64,
      nullptr, 0, 0, 0, H, 64);

  // --- reprojection + residual: x = f2 + acat @ rp_w + rp_b ---
  k_gemm<EP_F32_RES><<<dim3(D / 32, M2 / 32, 1), 32, 0, stream>>>(
      acat, 2 * D, 0, 0, rpT, 2 * D, 0, 0, rp_b,
      xbuf, nullptr, D, 0, 0, f2, D, 0, 0, 1, 2 * D);

  // --- LN4 -> bf16 ---
  k_ln<<<dim3(M2 / 8), 256, 0, stream>>>(xbuf, ln4_w, ln4_b, nullptr, nullptr, xln, N2, M2);

  // --- MLP: hh = gelu(xln @ ff1 + b); out = x + hh @ ff2 + b ---
  k_gemm<EP_BF16_GELU><<<dim3(MLP / 32, M2 / 32, 1), 32, 0, stream>>>(
      xln, D, 0, 0, ff1T, D, 0, 0, ff1_b,
      nullptr, hh, MLP, 0, 0, nullptr, 0, 0, 0, 1, D);
  k_gemm<EP_F32_RES><<<dim3(D / 32, M2 / 32, 1), 32, 0, stream>>>(
      hh, MLP, 0, 0, ff2T, MLP, 0, 0, ff2_b,
      (float*)d_out, nullptr, D, 0, 0, xbuf, D, 0, 0, 1, MLP);
}
